// DLRM_Net_13546326852241
// MI455X (gfx1250) — matrix operand order repaired and checked by reading, not compiled, not run on hardware
//
#include <hip/hip_runtime.h>
#include <stdint.h>

// ---------------- problem constants (match reference) ----------------
#define BB    16384      // batch
#define NTAB  26
#define NROWS 100000
#define DEMB  128
// bottom: 13 -> 512 -> 256 -> 128 ; top: 479 -> 1024 -> 512 -> 256 -> 1
// padded K's: 13->32, 479->512

typedef __attribute__((ext_vector_type(16))) __bf16 v16bf;
typedef __attribute__((ext_vector_type(8)))  float  v8f;

union FragB { uint4 u[2]; v16bf v; };

__device__ __forceinline__ unsigned short f2bf(float f) {
  unsigned u = __float_as_uint(f);
  unsigned r = u + 0x7FFFu + ((u >> 16) & 1u);   // RNE
  return (unsigned short)(r >> 16);
}
__device__ __forceinline__ float bf2f(unsigned short h) {
  return __uint_as_float(((unsigned)h) << 16);
}

// ---------------- convert f32 -> bf16 with K padding ----------------
__global__ void k_cvt_pad(const float* __restrict__ src, unsigned short* __restrict__ dst,
                          int rows, int K, int Kp) {
  long long n = (long long)rows * Kp;
  for (long long i = (long long)blockIdx.x * blockDim.x + threadIdx.x; i < n;
       i += (long long)gridDim.x * blockDim.x) {
    int r = (int)(i / Kp), k = (int)(i - (long long)r * Kp);
    dst[i] = (k < K) ? f2bf(src[(long long)r * K + k]) : (unsigned short)0;
  }
}

// ---------------- embedding gather + pool (P=2) -> T rows 1..26 -------
// one wave per (b, t); each lane moves one float4 per row (128 dims / 32 lanes)
__global__ void k_gather(const float* __restrict__ emb, const int* __restrict__ lsi,
                         unsigned short* __restrict__ T) {
  int wave = (int)(((long long)blockIdx.x * blockDim.x + threadIdx.x) >> 5);
  int lane = threadIdx.x & 31;
  if (wave >= BB * NTAB) return;
  int b = wave / NTAB, t = wave - b * NTAB;
  long long base = (long long)t * (BB * 2) + (long long)b * 2;
  long long i0 = lsi[base];
  long long i1 = lsi[base + 1];
  const float4* r0 = (const float4*)(emb + ((long long)t * NROWS + i0) * DEMB) + lane;
  const float4* r1 = (const float4*)(emb + ((long long)t * NROWS + i1) * DEMB) + lane;
  __builtin_prefetch(r1, 0, 1);          // global_prefetch_b8 on 2nd pooled row
  float4 a = *r0;
  float4 c = *r1;
  unsigned short h0 = f2bf(a.x + c.x), h1 = f2bf(a.y + c.y);
  unsigned short h2 = f2bf(a.z + c.z), h3 = f2bf(a.w + c.w);
  uint2 pk;
  pk.x = (unsigned)h0 | ((unsigned)h1 << 16);
  pk.y = (unsigned)h2 | ((unsigned)h3 << 16);
  *(uint2*)(T + ((long long)b * 32 + 1 + t) * DEMB + lane * 4) = pk;
}

// ---------------- bf16 WMMA GEMM: out = relu(A @ W^T + b) ----------------
// A: M x Kp (row major bf16), W: N x Kp (row major bf16), out: M x N bf16
// one wave per 32x32 C block (2x2 register tiling): 4 independent
// v_wmma_f32_16x16x32_bf16 per 32-K step, A/B fragments each reused twice.
__global__ void k_gemm(const unsigned short* __restrict__ A,
                       const unsigned short* __restrict__ W,
                       const float* __restrict__ bias,
                       unsigned short* __restrict__ out,
                       int M, int N, int Kp, int relu) {
  int wave = (int)(((long long)blockIdx.x * blockDim.x + threadIdx.x) >> 5);
  int lane = threadIdx.x & 31;
  int blocksN = N >> 5;
  int bM = wave / blocksN, bN = wave - bM * blocksN;
  if (bM >= (M >> 5)) return;
  int m0 = bM << 5, n0 = bN << 5;
  int hl  = lane >> 4;        // half-wave select
  int l15 = lane & 15;
  const unsigned short* arow0 = A + (long long)(m0 + l15) * Kp;
  const unsigned short* arow1 = arow0 + (long long)16 * Kp;
  const unsigned short* wrow0 = W + (long long)(n0 + l15) * Kp;
  const unsigned short* wrow1 = wrow0 + (long long)16 * Kp;
  v8f acc00 = {}, acc01 = {}, acc10 = {}, acc11 = {};
  for (int k0 = 0; k0 < Kp; k0 += 32) {
    FragB fa0, fa1, fb0, fb1;
    // A 16x32 bf16 layout: half h -> K[8h..8h+7] in v0-3, K[16+8h..23+8h] in v4-7
    fa0.u[0] = *(const uint4*)(arow0 + k0 + 8 * hl);
    fa0.u[1] = *(const uint4*)(arow0 + k0 + 16 + 8 * hl);
    fa1.u[0] = *(const uint4*)(arow1 + k0 + 8 * hl);
    fa1.u[1] = *(const uint4*)(arow1 + k0 + 16 + 8 * hl);
    // B 32x16 bf16 layout: lane n=l15, K[8h..8h+7] in v0-3, K[16+8h..23+8h] in v4-7
    fb0.u[0] = *(const uint4*)(wrow0 + k0 + 8 * hl);
    fb0.u[1] = *(const uint4*)(wrow0 + k0 + 16 + 8 * hl);
    fb1.u[0] = *(const uint4*)(wrow1 + k0 + 8 * hl);
    fb1.u[1] = *(const uint4*)(wrow1 + k0 + 16 + 8 * hl);
    acc00 = __builtin_amdgcn_wmma_f32_16x16x32_bf16(false, fa0.v, false, fb0.v,
                                                    (short)0, acc00, false, false);
    acc01 = __builtin_amdgcn_wmma_f32_16x16x32_bf16(false, fa0.v, false, fb1.v,
                                                    (short)0, acc01, false, false);
    acc10 = __builtin_amdgcn_wmma_f32_16x16x32_bf16(false, fa1.v, false, fb0.v,
                                                    (short)0, acc10, false, false);
    acc11 = __builtin_amdgcn_wmma_f32_16x16x32_bf16(false, fa1.v, false, fb1.v,
                                                    (short)0, acc11, false, false);
  }
  float bv0 = bias[n0 + l15];
  float bv1 = bias[n0 + 16 + l15];
  // C layout: VGPR r -> M = r + 8*half, N = lane&15 (within tile)
#pragma unroll
  for (int r = 0; r < 8; ++r) {
    int mA = m0 + r + 8 * hl;
    int mB = mA + 16;
    int nA = n0 + l15;
    int nB = nA + 16;
    float v00 = acc00[r] + bv0, v01 = acc01[r] + bv1;
    float v10 = acc10[r] + bv0, v11 = acc11[r] + bv1;
    if (relu) {
      v00 = v00 > 0.f ? v00 : 0.f;
      v01 = v01 > 0.f ? v01 : 0.f;
      v10 = v10 > 0.f ? v10 : 0.f;
      v11 = v11 > 0.f ? v11 : 0.f;
    }
    out[(long long)mA * N + nA] = f2bf(v00);
    out[(long long)mA * N + nB] = f2bf(v01);
    out[(long long)mB * N + nA] = f2bf(v10);
    out[(long long)mB * N + nB] = f2bf(v11);
  }
}

// ---------------- scatter x into T row0 + R[:, :128] ----------------
__global__ void k_copy_x(const unsigned short* __restrict__ X,
                         unsigned short* __restrict__ T,
                         unsigned short* __restrict__ R) {
  long long n = (long long)BB * DEMB;
  for (long long i = (long long)blockIdx.x * blockDim.x + threadIdx.x; i < n;
       i += (long long)gridDim.x * blockDim.x) {
    long long b = i >> 7;
    int d = (int)(i & 127);
    unsigned short v = X[i];
    T[b * (32 * DEMB) + d] = v;
    R[b * 512 + d] = v;
  }
}

// zero R cols 479..511 (pad region hit by tw0's zero-padded K)
__global__ void k_zero_rpad(unsigned short* __restrict__ R) {
  long long n = (long long)BB * 33;
  for (long long i = (long long)blockIdx.x * blockDim.x + threadIdx.x; i < n;
       i += (long long)gridDim.x * blockDim.x) {
    long long b = i / 33;
    int j = (int)(i - b * 33);
    R[b * 512 + 479 + j] = 0;
  }
}

// ---------------- interaction: Z = T T^T per sample, tril -> R --------
// one wave per sample; 3 tiles of the 32x32 (pad rows 27..31 discarded)
__global__ void k_interact(const unsigned short* __restrict__ T,
                           unsigned short* __restrict__ R) {
  int wave = (int)(((long long)blockIdx.x * blockDim.x + threadIdx.x) >> 5);
  int lane = threadIdx.x & 31;
  if (wave >= BB) return;
  const unsigned short* Tb = T + (long long)wave * 32 * DEMB;
  unsigned short* Rb = R + (long long)wave * 512 + 128;
  int hl = lane >> 4, l15 = lane & 15;
  const int ti[3] = {0, 1, 1};
  const int tj[3] = {0, 0, 1};
#pragma unroll
  for (int t = 0; t < 3; ++t) {
    int m0 = ti[t] << 4, n0 = tj[t] << 4;
    const unsigned short* arow = Tb + (m0 + l15) * DEMB;
    const unsigned short* brow = Tb + (n0 + l15) * DEMB;   // B[k][n] = T[n][k]
    v8f acc = {};
#pragma unroll
    for (int k0 = 0; k0 < DEMB; k0 += 32) {
      FragB fa, fb;
      fa.u[0] = *(const uint4*)(arow + k0 + 8 * hl);
      fa.u[1] = *(const uint4*)(arow + k0 + 16 + 8 * hl);
      fb.u[0] = *(const uint4*)(brow + k0 + 8 * hl);
      fb.u[1] = *(const uint4*)(brow + k0 + 16 + 8 * hl);
      acc = __builtin_amdgcn_wmma_f32_16x16x32_bf16(false, fa.v, false, fb.v,
                                                    (short)0, acc, false, false);
    }
    int n = n0 + l15;
#pragma unroll
    for (int r = 0; r < 8; ++r) {
      int m = m0 + r + 8 * hl;
      if (m > n && m <= 26) {               // strict lower triangle, real rows only
        Rb[(m * (m - 1)) / 2 + n] = f2bf(acc[r]);
      }
    }
  }
}

// ---------------- final 256 -> 1 + sigmoid ----------------
__global__ void k_final(const unsigned short* __restrict__ Y2,
                        const float* __restrict__ w, const float* __restrict__ b,
                        float* __restrict__ out) {
  int wave = (int)(((long long)blockIdx.x * blockDim.x + threadIdx.x) >> 5);
  int lane = threadIdx.x & 31;
  if (wave >= BB) return;
  const unsigned short* y = Y2 + (long long)wave * 256;
  float s = 0.f;
#pragma unroll
  for (int k = lane; k < 256; k += 32) s += bf2f(y[k]) * w[k];
#pragma unroll
  for (int off = 16; off; off >>= 1) s += __shfl_xor(s, off, 32);
  if (lane == 0) out[wave] = 1.f / (1.f + __expf(-(s + b[0])));
}

// ---------------- host side ----------------
static inline void launch_cvt(const float* s, unsigned short* d, int rows, int K, int Kp,
                              hipStream_t st) {
  long long n = (long long)rows * Kp;
  int blocks = (int)((n + 255) / 256);
  if (blocks > 16384) blocks = 16384;
  k_cvt_pad<<<blocks, 256, 0, st>>>(s, d, rows, K, Kp);
}
static inline void launch_gemm(const unsigned short* A, const unsigned short* W,
                               const float* bias, unsigned short* out,
                               int M, int N, int Kp, int relu, hipStream_t st) {
  int waves = (M >> 5) * (N >> 5);   // one wave per 32x32 C block
  k_gemm<<<waves / 8, 256, 0, st>>>(A, W, bias, out, M, N, Kp, relu);
}

extern "C" void kernel_launch(void* const* d_in, const int* in_sizes, int n_in,
                              void* d_out, int out_size, void* d_ws, size_t ws_size,
                              hipStream_t stream) {
  const float* dense_x = (const float*)d_in[0];
  const int*   lS_i    = (const int*)d_in[2];          // lS_o (d_in[1]) is b*P, unused
  const float* emb     = (const float*)d_in[3];
  const float* bw0 = (const float*)d_in[4];  const float* bb0 = (const float*)d_in[5];
  const float* bw1 = (const float*)d_in[6];  const float* bb1 = (const float*)d_in[7];
  const float* bw2 = (const float*)d_in[8];  const float* bb2 = (const float*)d_in[9];
  const float* tw0 = (const float*)d_in[10]; const float* tb0 = (const float*)d_in[11];
  const float* tw1 = (const float*)d_in[12]; const float* tb1 = (const float*)d_in[13];
  const float* tw2 = (const float*)d_in[14]; const float* tb2 = (const float*)d_in[15];
  const float* tw3 = (const float*)d_in[16]; const float* tb3 = (const float*)d_in[17];
  float* out = (float*)d_out;

  // carve workspace (256-B aligned regions)
  char* ws = (char*)d_ws;
  size_t off = 0;
  auto carve = [&](size_t bytes) -> unsigned short* {
    unsigned short* p = (unsigned short*)(ws + off);
    off += (bytes + 255) & ~(size_t)255;
    return p;
  };
  unsigned short* X0p = carve((size_t)BB * 32 * 2);        // padded dense_x
  unsigned short* Wb0 = carve((size_t)512 * 32 * 2);
  unsigned short* Wb1 = carve((size_t)256 * 512 * 2);
  unsigned short* Wb2 = carve((size_t)128 * 256 * 2);
  unsigned short* Wt0 = carve((size_t)1024 * 512 * 2);     // tw0 padded 479->512
  unsigned short* Wt1 = carve((size_t)512 * 1024 * 2);
  unsigned short* Wt2 = carve((size_t)256 * 512 * 2);
  unsigned short* A0  = carve((size_t)BB * 512 * 2);
  unsigned short* A1  = carve((size_t)BB * 256 * 2);
  unsigned short* Xb  = carve((size_t)BB * 128 * 2);       // bottom output x
  unsigned short* T   = carve((size_t)BB * 32 * DEMB * 2); // [x ; 26 pooled ; 5 pad]
  unsigned short* R   = carve((size_t)BB * 512 * 2);       // [x(128) Z(351) pad(33)]
  unsigned short* Y0  = carve((size_t)BB * 1024 * 2);
  unsigned short* Y1  = carve((size_t)BB * 512 * 2);
  unsigned short* Y2  = carve((size_t)BB * 256 * 2);
  (void)ws_size; (void)in_sizes; (void)n_in; (void)out_size;

  // weight / input conversion (bf16, K padded to mult of 32)
  launch_cvt(dense_x, X0p, BB, 13, 32, stream);
  launch_cvt(bw0, Wb0, 512, 13, 32, stream);
  launch_cvt(bw1, Wb1, 256, 512, 512, stream);
  launch_cvt(bw2, Wb2, 128, 256, 256, stream);
  launch_cvt(tw0, Wt0, 1024, 479, 512, stream);
  launch_cvt(tw1, Wt1, 512, 1024, 1024, stream);
  launch_cvt(tw2, Wt2, 256, 512, 512, stream);

  // embedding gather+pool (the bandwidth-dominant step)
  {
    int waves = BB * NTAB;
    k_gather<<<waves / 8, 256, 0, stream>>>(emb, lS_i, T);
  }

  // bottom MLP
  launch_gemm(X0p, Wb0, bb0, A0, BB, 512, 32, 1, stream);
  launch_gemm(A0,  Wb1, bb1, A1, BB, 256, 512, 1, stream);
  launch_gemm(A1,  Wb2, bb2, Xb, BB, 128, 256, 1, stream);

  // assemble T row0 and R
  {
    long long n = (long long)BB * DEMB;
    k_copy_x<<<(int)((n + 255) / 256), 256, 0, stream>>>(Xb, T, R);
    long long nz = (long long)BB * 33;
    k_zero_rpad<<<(int)((nz + 255) / 256), 256, 0, stream>>>(R);
  }

  // pairwise interaction -> R[:, 128:479]
  k_interact<<<BB / 8, 256, 0, stream>>>(T, R);

  // top MLP
  launch_gemm(R,  Wt0, tb0, Y0, BB, 1024, 512, 1, stream);
  launch_gemm(Y0, Wt1, tb1, Y1, BB, 512, 1024, 1, stream);
  launch_gemm(Y1, Wt2, tb2, Y2, BB, 256, 512, 1, stream);

  // final 256->1 + sigmoid
  k_final<<<BB / 8, 256, 0, stream>>>(Y2, tw3, tb3, out);
}
